// SAGE_65970697666938
// MI455X (gfx1250) — matrix operand order repaired and checked by reading, not compiled, not run on hardware
//
#include <hip/hip_runtime.h>
#include <stdint.h>

// ---------------- problem constants (from reference) ----------------
#define DIM       128
#define FANOUT    5
#define NUM_EDGE  20000
#define NEG       1
#define NUM_DST1  ((NEG + 2) * NUM_EDGE)          // 60000
#define NUM_DST0  (NUM_DST1 * (1 + FANOUT))       // 360000
#define NUM_SRC0  (NUM_DST0 * (1 + FANOUT))       // 2160000
#define PRED_ROWS (2 * NUM_EDGE)                  // 40000 (pos + neg)

// ---------------- workspace layout (uint16_t element offsets) --------
static constexpr size_t WCAT0_OFF = 0;                        // 128x256 bf16
static constexpr size_t WCAT1_OFF = WCAT0_OFF + 128 * 256;    // 128x256 bf16
static constexpr size_t WP1_OFF   = WCAT1_OFF + 128 * 256;    // 128x128 bf16
static constexpr size_t WP2_OFF   = WP1_OFF + 128 * 128;      // 128x128 bf16
static constexpr size_t WP3_OFF   = WP2_OFF + 128 * 128;      // 128 bf16
static constexpr size_t H0_OFF    = 131072;                           // 360000x128 bf16
static constexpr size_t H1_OFF    = H0_OFF + (size_t)NUM_DST0 * DIM;  // 60000x128 bf16
static constexpr size_t Z_OFF     = H1_OFF + (size_t)NUM_DST1 * DIM;  // 40000x128 bf16

// ---------------- types / helpers ----------------
typedef __attribute__((ext_vector_type(16))) __bf16    v16bf;
typedef __attribute__((ext_vector_type(8)))  float     v8f;
typedef __attribute__((ext_vector_type(4)))  float     f32x4;
typedef __attribute__((ext_vector_type(2)))  uint32_t  u32x2;

__device__ __forceinline__ uint16_t f32_to_bf16(float f) {
  uint32_t u = __builtin_bit_cast(uint32_t, f);
  u += 0x7FFFu + ((u >> 16) & 1u);           // round-to-nearest-even
  return (uint16_t)(u >> 16);
}
__device__ __forceinline__ float bf16_to_f32(uint16_t h) {
  uint32_t u = ((uint32_t)h) << 16;
  return __builtin_bit_cast(float, u);
}

union FragBF { uint32_t u[8]; v16bf v; };

// ---------------- kernel 1: weight conversion / packing ----------------
__global__ void convert_weights_kernel(const float* __restrict__ Wself0,
                                       const float* __restrict__ Wneigh0,
                                       const float* __restrict__ Wself1,
                                       const float* __restrict__ Wneigh1,
                                       const float* __restrict__ Wp1,
                                       const float* __restrict__ Wp2,
                                       const float* __restrict__ Wp3,
                                       uint16_t* __restrict__ ws) {
  int idx = blockIdx.x * 256 + threadIdx.x;
  if (idx < 32768) {                                   // Wcat0[n][k], k<128:self else neigh
    int n = idx >> 8, k = idx & 255;
    float v = (k < 128) ? Wself0[n * 128 + k] : Wneigh0[n * 128 + (k - 128)];
    ws[WCAT0_OFF + idx] = f32_to_bf16(v);
  } else if (idx < 65536) {
    int j = idx - 32768;
    int n = j >> 8, k = j & 255;
    float v = (k < 128) ? Wself1[n * 128 + k] : Wneigh1[n * 128 + (k - 128)];
    ws[WCAT1_OFF + j] = f32_to_bf16(v);
  } else if (idx < 81920) {
    int j = idx - 65536;
    ws[WP1_OFF + j] = f32_to_bf16(Wp1[j]);
  } else if (idx < 98304) {
    int j = idx - 81920;
    ws[WP2_OFF + j] = f32_to_bf16(Wp2[j]);
  } else if (idx < 98432) {
    int j = idx - 98304;
    ws[WP3_OFF + j] = f32_to_bf16(Wp3[j]);
  }
}

// ---------------- kernel 2/3: fused SAGE conv (gather-mean + GEMM) ------
// One wave computes a 16x128 output tile; A = [self | mean-neigh] (16x256 bf16),
// B = Wcat (bf16, row-major [n][256] in workspace, read as packed u32; stays
// resident in WGP$/L2 because the streaming feature gather uses TH=NT loads).
template <bool FEAT_BF16, bool RELU>
__global__ __launch_bounds__(128) void sage_kernel(const void* __restrict__ feat_,
                                                   const int* __restrict__ src,
                                                   const uint32_t* __restrict__ wcat,
                                                   const float* __restrict__ bias,
                                                   int num_dst,
                                                   uint16_t* __restrict__ out) {
  __shared__ uint16_t sA[4][16 * 256];   // 32 KB: per-wave 16x256 bf16 A tile
  const int wave = threadIdx.x >> 5;
  const int lane = threadIdx.x & 31;
  const int rowbase = blockIdx.x * 64 + wave * 16;
  const int c0 = lane * 4;               // 32 lanes x 4 cols = 128

  // ---- gather + mean into LDS (bf16); feature rows are non-temporal ----
  for (int m = 0; m < 16; ++m) {
    const int r = rowbase + m;
    float sv[4] = {0.f, 0.f, 0.f, 0.f};
    float nb[4] = {0.f, 0.f, 0.f, 0.f};
    if (r < num_dst) {
      if (FEAT_BF16) {
        const uint16_t* f = (const uint16_t*)feat_;
        {
          u32x2 p = __builtin_nontemporal_load((const u32x2*)(f + (size_t)r * DIM + c0));
          sv[0] = bf16_to_f32((uint16_t)(p.x & 0xffff)); sv[1] = bf16_to_f32((uint16_t)(p.x >> 16));
          sv[2] = bf16_to_f32((uint16_t)(p.y & 0xffff)); sv[3] = bf16_to_f32((uint16_t)(p.y >> 16));
        }
        #pragma unroll
        for (int j = 0; j < FANOUT; ++j) {
          const int s = src[r * FANOUT + j];
          u32x2 p = __builtin_nontemporal_load((const u32x2*)(f + (size_t)s * DIM + c0));
          nb[0] += bf16_to_f32((uint16_t)(p.x & 0xffff)); nb[1] += bf16_to_f32((uint16_t)(p.x >> 16));
          nb[2] += bf16_to_f32((uint16_t)(p.y & 0xffff)); nb[3] += bf16_to_f32((uint16_t)(p.y >> 16));
        }
      } else {
        const float* f = (const float*)feat_;
        const f32x4 s4 = __builtin_nontemporal_load((const f32x4*)(f + (size_t)r * DIM + c0));
        sv[0] = s4.x; sv[1] = s4.y; sv[2] = s4.z; sv[3] = s4.w;
        #pragma unroll
        for (int j = 0; j < FANOUT; ++j) {
          const int s = src[r * FANOUT + j];
          const f32x4 n4 = __builtin_nontemporal_load((const f32x4*)(f + (size_t)s * DIM + c0));
          nb[0] += n4.x; nb[1] += n4.y; nb[2] += n4.z; nb[3] += n4.w;
        }
      }
    }
    #pragma unroll
    for (int t = 0; t < 4; ++t) {
      sA[wave][m * 256 + c0 + t]       = f32_to_bf16(sv[t]);
      sA[wave][m * 256 + 128 + c0 + t] = f32_to_bf16(nb[t] * 0.2f);   // /FANOUT
    }
  }
  __syncthreads();

  // ---- WMMA GEMM: (16x256) x (256x128) -> 16x128 ----
  const int m_l = lane & 15;
  const int hi  = lane >> 4;
  const uint32_t* sA32 = (const uint32_t*)&sA[wave][0];

  v8f acc[8];
  const v8f zed = {};
  #pragma unroll
  for (int nt = 0; nt < 8; ++nt) acc[nt] = zed;

  #pragma unroll
  for (int kt = 0; kt < 8; ++kt) {                 // K = 256, 32 per step
    FragBF af;
    #pragma unroll
    for (int vv = 0; vv < 8; ++vv) {
      const int kk = (vv < 4 ? 2 * vv : 16 + 2 * (vv - 4)) + 8 * hi;   // A layout
      af.u[vv] = sA32[(m_l * 256 + kt * 32 + kk) >> 1];
    }
    #pragma unroll
    for (int nt = 0; nt < 8; ++nt) {
      const int n = nt * 16 + m_l;
      FragBF bfm;
      #pragma unroll
      for (int vv = 0; vv < 8; ++vv) {
        bfm.u[vv] = wcat[n * 128 + kt * 16 + (vv < 4 ? vv : 4 + vv) + 4 * hi];             // B layout: k=2v+8*half (+8 for v>=4)
      }
      acc[nt] = __builtin_amdgcn_wmma_f32_16x16x32_bf16(
          false, af.v, false, bfm.v, (short)0, acc[nt], false, false);
    }
  }

  // ---- epilogue: bias (+ReLU), store bf16 ----
  #pragma unroll
  for (int nt = 0; nt < 8; ++nt) {
    const int n = nt * 16 + m_l;
    const float bv = bias[n];
    #pragma unroll
    for (int vv = 0; vv < 8; ++vv) {
      const int m = vv + 8 * hi;
      const int r = rowbase + m;
      if (r < num_dst) {
        float v = acc[nt][vv] + bv;
        if (RELU) v = fmaxf(v, 0.0f);
        out[(size_t)r * DIM + n] = f32_to_bf16(v);
      }
    }
  }
}

// ---------------- kernel 4: z = src_h * dst_h (pos & neg) ----------------
__global__ void zmul_kernel(const uint16_t* __restrict__ h1, uint16_t* __restrict__ z) {
  const int total = 2 * NUM_EDGE * DIM;
  int idx = blockIdx.x * 256 + threadIdx.x;
  if (idx >= total) return;
  const int which = idx / (NUM_EDGE * DIM);        // 0 = pos, 1 = neg
  const int rem   = idx - which * (NUM_EDGE * DIM);
  const int e = rem >> 7, c = rem & 127;
  const float a = bf16_to_f32(h1[(size_t)e * DIM + c]);
  const float b = bf16_to_f32(h1[(size_t)(e + (which ? 2 : 1) * NUM_EDGE) * DIM + c]);
  z[idx] = f32_to_bf16(a * b);
}

// ---------------- kernel 5: fused 3-layer predictor ----------------
// z-tile staging uses the gfx1250 async global->LDS DMA path (ASYNCcnt).
__global__ __launch_bounds__(128) void mlp_kernel(const uint16_t* __restrict__ z,
                                                  const uint32_t* __restrict__ wp1,
                                                  const uint32_t* __restrict__ wp2,
                                                  const uint16_t* __restrict__ wp3,
                                                  const float* __restrict__ bp1,
                                                  const float* __restrict__ bp2,
                                                  const float* __restrict__ bp3,
                                                  float* __restrict__ out) {
  __shared__ uint16_t sA[4][16 * 128];   // 16 KB: activation tiles (bf16)
  __shared__ float    sH2[4][16 * 128];  // 32 KB: layer-2 output (f32)
  const int wave = threadIdx.x >> 5;
  const int lane = threadIdx.x & 31;
  const int rowbase = blockIdx.x * 64 + wave * 16;   // PRED_ROWS % 64 == 0: no guards
  const int m_l = lane & 15;
  const int hi  = lane >> 4;
  const v8f zed = {};

  // ---- async copy z tile (16 rows x 256 B) global -> LDS ----
  {
    const uint32_t lds_base = (uint32_t)(uintptr_t)(&sA[wave][0]);
    const uint64_t g_base   = (uint64_t)(uintptr_t)(z + (size_t)rowbase * DIM);
    #pragma unroll
    for (int i = 0; i < 8; ++i) {
      const uint32_t byteoff = (uint32_t)(i * 512 + lane * 16);  // 32 lanes x 16 B
      const uint32_t laddr = lds_base + byteoff;
      const uint64_t gaddr = g_base + byteoff;                   // tile is contiguous
      asm volatile("global_load_async_to_lds_b128 %0, %1, off"
                   :: "v"(laddr), "v"(gaddr) : "memory");
    }
    asm volatile("s_wait_asynccnt 0x0" ::: "memory");
  }
  __syncthreads();
  const uint32_t* sA32 = (const uint32_t*)&sA[wave][0];

  // ---- GEMM1: relu(z @ Wp1.T + bp1) ----
  v8f acc[8];
  #pragma unroll
  for (int nt = 0; nt < 8; ++nt) acc[nt] = zed;
  #pragma unroll
  for (int kt = 0; kt < 4; ++kt) {                 // K = 128
    FragBF af;
    #pragma unroll
    for (int vv = 0; vv < 8; ++vv) {
      const int kk = (vv < 4 ? 2 * vv : 16 + 2 * (vv - 4)) + 8 * hi;
      af.u[vv] = sA32[(m_l * 128 + kt * 32 + kk) >> 1];
    }
    #pragma unroll
    for (int nt = 0; nt < 8; ++nt) {
      const int n = nt * 16 + m_l;
      FragBF bfm;
      #pragma unroll
      for (int vv = 0; vv < 8; ++vv) bfm.u[vv] = wp1[n * 64 + kt * 16 + (vv < 4 ? vv : 4 + vv) + 4 * hi];
      acc[nt] = __builtin_amdgcn_wmma_f32_16x16x32_bf16(
          false, af.v, false, bfm.v, (short)0, acc[nt], false, false);
    }
  }
  __syncthreads();
  #pragma unroll
  for (int nt = 0; nt < 8; ++nt) {
    const int n = nt * 16 + m_l;
    const float bv = bp1[n];
    #pragma unroll
    for (int vv = 0; vv < 8; ++vv) {
      const int m = vv + 8 * hi;
      sA[wave][m * 128 + n] = f32_to_bf16(fmaxf(acc[nt][vv] + bv, 0.0f));
    }
  }
  __syncthreads();

  // ---- GEMM2: relu(h @ Wp2.T + bp2) -> sH2 (f32) ----
  #pragma unroll
  for (int nt = 0; nt < 8; ++nt) acc[nt] = zed;
  #pragma unroll
  for (int kt = 0; kt < 4; ++kt) {
    FragBF af;
    #pragma unroll
    for (int vv = 0; vv < 8; ++vv) {
      const int kk = (vv < 4 ? 2 * vv : 16 + 2 * (vv - 4)) + 8 * hi;
      af.u[vv] = sA32[(m_l * 128 + kt * 32 + kk) >> 1];
    }
    #pragma unroll
    for (int nt = 0; nt < 8; ++nt) {
      const int n = nt * 16 + m_l;
      FragBF bfm;
      #pragma unroll
      for (int vv = 0; vv < 8; ++vv) bfm.u[vv] = wp2[n * 64 + kt * 16 + (vv < 4 ? vv : 4 + vv) + 4 * hi];
      acc[nt] = __builtin_amdgcn_wmma_f32_16x16x32_bf16(
          false, af.v, false, bfm.v, (short)0, acc[nt], false, false);
    }
  }
  #pragma unroll
  for (int nt = 0; nt < 8; ++nt) {
    const int n = nt * 16 + m_l;
    const float bv = bp2[n];
    #pragma unroll
    for (int vv = 0; vv < 8; ++vv) {
      const int m = vv + 8 * hi;
      sH2[wave][m * 128 + n] = fmaxf(acc[nt][vv] + bv, 0.0f);
    }
  }
  __syncthreads();

  // ---- final row-dot with Wp3 (128 -> 1) ----
  if (lane < 16) {
    const int r = rowbase + lane;
    float s = bp3[0];
    #pragma unroll 4
    for (int c = 0; c < 128; ++c) {
      s += sH2[wave][lane * 128 + c] * bf16_to_f32(wp3[c]);
    }
    out[r] = s;   // rows [0,20000) = h_pos, rows [20000,40000) = h_neg
  }
}

// ---------------- host launcher ----------------
extern "C" void kernel_launch(void* const* d_in, const int* in_sizes, int n_in,
                              void* d_out, int out_size, void* d_ws, size_t ws_size,
                              hipStream_t stream) {
  (void)in_sizes; (void)n_in; (void)out_size; (void)ws_size;
  const float* x       = (const float*)d_in[0];
  const float* Wself0  = (const float*)d_in[1];
  const float* Wneigh0 = (const float*)d_in[2];
  const float* b0      = (const float*)d_in[3];
  const float* Wself1  = (const float*)d_in[4];
  const float* Wneigh1 = (const float*)d_in[5];
  const float* b1      = (const float*)d_in[6];
  const float* Wp1     = (const float*)d_in[7];
  const float* bp1     = (const float*)d_in[8];
  const float* Wp2     = (const float*)d_in[9];
  const float* bp2     = (const float*)d_in[10];
  const float* Wp3     = (const float*)d_in[11];
  const float* bp3     = (const float*)d_in[12];
  const int*   src0    = (const int*)d_in[13];
  const int*   src1    = (const int*)d_in[15];
  float* out = (float*)d_out;

  uint16_t* wsb = (uint16_t*)d_ws;
  const uint32_t* wcat0 = (const uint32_t*)(wsb + WCAT0_OFF);
  const uint32_t* wcat1 = (const uint32_t*)(wsb + WCAT1_OFF);
  const uint32_t* wp1p  = (const uint32_t*)(wsb + WP1_OFF);
  const uint32_t* wp2p  = (const uint32_t*)(wsb + WP2_OFF);
  const uint16_t* wp3p  = wsb + WP3_OFF;
  uint16_t* H0 = wsb + H0_OFF;
  uint16_t* H1 = wsb + H1_OFF;
  uint16_t* Z  = wsb + Z_OFF;

  // 1) pack weights to bf16
  convert_weights_kernel<<<385, 256, 0, stream>>>(Wself0, Wneigh0, Wself1, Wneigh1,
                                                  Wp1, Wp2, Wp3, wsb);
  // 2) layer 0: relu(sage(x)) -> H0 (bf16)
  sage_kernel<false, true><<<NUM_DST0 / 64, 128, 0, stream>>>(
      (const void*)x, src0, wcat0, b0, NUM_DST0, H0);
  // 3) layer 1: sage(H0) -> H1 (bf16, no relu)
  sage_kernel<true, false><<<(NUM_DST1 + 63) / 64, 128, 0, stream>>>(
      (const void*)H0, src1, wcat1, b1, NUM_DST1, H1);
  // 4) z = src_h * {pos,neg}_dst_h -> Z (bf16, 40000x128)
  zmul_kernel<<<(2 * NUM_EDGE * DIM + 255) / 256, 256, 0, stream>>>(H1, Z);
  // 5) fused predictor -> d_out (40000 f32: h_pos then h_neg)
  mlp_kernel<<<PRED_ROWS / 64, 128, 0, stream>>>(Z, wp1p, wp2p, wp3p, bp1, bp2, bp3, out);
}
